// GATEncoderWrapper_11132555231669
// MI455X (gfx1250) — compile-verified, matrix operand order checked by reading, not run on hardware
//
#include <hip/hip_runtime.h>

#define WPB 8  // waves per block (256 threads, wave32)

typedef __attribute__((ext_vector_type(16))) __bf16 bf16x16;
typedef __attribute__((ext_vector_type(8)))  float  f32x8;

union Frag { bf16x16 f; uint4 u[2]; };

__device__ __forceinline__ unsigned short f2bf(float x) {
  unsigned b = __float_as_uint(x);
  b += 0x7FFFu + ((b >> 16) & 1u);   // round-to-nearest-even
  return (unsigned short)(b >> 16);
}
__device__ __forceinline__ unsigned encf(float f) {   // monotone float->uint
  unsigned b = __float_as_uint(f);
  return (b & 0x80000000u) ? ~b : (b | 0x80000000u);
}
__device__ __forceinline__ float decf(unsigned u) {
  unsigned b = (u & 0x80000000u) ? (u & 0x7FFFFFFFu) : ~u;
  return __uint_as_float(b);
}

// ---- elementwise helpers -------------------------------------------------
// dst is [rows_total, Kpad] bf16; source valid region is [rows_src, K] f32.
__global__ void k_convert_pad(const float* __restrict__ src, unsigned short* __restrict__ dst,
                              int rows_total, int rows_src, int K, int Kpad) {
  int i = blockIdx.x * blockDim.x + threadIdx.x;
  if (i >= rows_total * Kpad) return;
  int r = i / Kpad, k = i - r * Kpad;
  dst[i] = (r < rows_src && k < K) ? f2bf(src[(size_t)r * K + k]) : (unsigned short)0;
}
__global__ void k_fill_f32(float* p, float v, int n) {
  int i = blockIdx.x * blockDim.x + threadIdx.x;
  if (i < n) p[i] = v;
}
__global__ void k_fill_u32(unsigned* p, unsigned v, int n) {
  int i = blockIdx.x * blockDim.x + threadIdx.x;
  if (i < n) p[i] = v;
}
__global__ void k_bias_init(float* out, const float* __restrict__ bias, int n, int ncols) {
  int i = blockIdx.x * blockDim.x + threadIdx.x;
  if (i < n) out[i] = bias[i % ncols];
}
// dst[i] = bf16(elu(acc[i]+bias)) for i<nvalid, else 0 (zeroes the M-pad rows)
__global__ void k_post_elu(const float* __restrict__ acc, const float* __restrict__ bias,
                           unsigned short* __restrict__ dst, int nvalid, int ntotal, int ncols) {
  int i = blockIdx.x * blockDim.x + threadIdx.x;
  if (i >= ntotal) return;
  float v = 0.f;
  if (i < nvalid) {
    v = acc[i] + bias[i % ncols];
    v = v > 0.f ? v : (expf(v) - 1.f);
  }
  dst[i] = f2bf(v);
}

// ---- WMMA GEMM: out[Mpad,Ncols] = A[Mpad,K](bf16) * B[Ncols,K](bf16)^T + bias
// One wave computes a 32x64 strip: 2 A-frags x 4 B-frags -> 8 v_wmma per k-step
// (12 b128 loads per 8 WMMAs = 1.5 loads/wmma).
__global__ void k_gemm_wmma(const unsigned short* __restrict__ A,
                            const unsigned short* __restrict__ Bl,
                            const unsigned short* __restrict__ Br,
                            const float* __restrict__ biasl,
                            const float* __restrict__ biasr,
                            float* __restrict__ outl,
                            float* __restrict__ outr,
                            int K, int Ncols, int mtiles /* of 32 rows */) {
  int wid  = blockIdx.x * WPB + (int)(threadIdx.x >> 5);
  int lane = (int)(threadIdx.x & 31);
  int ngroups = Ncols >> 6;
  int perMat  = mtiles * ngroups;
  if (wid >= 2 * perMat) return;          // whole-wave uniform exit
  int lr  = wid >= perMat;
  int rem = lr ? wid - perMat : wid;
  int mt  = rem / ngroups;
  int ng  = rem - mt * ngroups;
  const unsigned short* B = lr ? Br : Bl;
  const float* bias = lr ? biasr : biasl;
  float* out = lr ? outr : outl;
  int m0 = mt << 5, n0 = ng << 6;
  int half = (lane >> 4) << 3;            // lanes 16-31 hold the K+8 halves
  const unsigned short* pa0 = A + (size_t)(m0 + (lane & 15)) * K + half;
  const unsigned short* pa1 = pa0 + (size_t)16 * K;
  const unsigned short* pb  = B + (size_t)(n0 + (lane & 15)) * K + half;
  const size_t bstep = (size_t)16 * K;    // 16 output-channel rows
  f32x8 c00 = {}, c01 = {}, c02 = {}, c03 = {};
  f32x8 c10 = {}, c11 = {}, c12 = {}, c13 = {};
  for (int k0 = 0; k0 < K; k0 += 32) {
    Frag a0, a1, b0, b1, b2, b3;
    a0.u[0] = *(const uint4*)(pa0);             a0.u[1] = *(const uint4*)(pa0 + 16);
    a1.u[0] = *(const uint4*)(pa1);             a1.u[1] = *(const uint4*)(pa1 + 16);
    b0.u[0] = *(const uint4*)(pb);              b0.u[1] = *(const uint4*)(pb + 16);
    b1.u[0] = *(const uint4*)(pb + bstep);      b1.u[1] = *(const uint4*)(pb + bstep + 16);
    b2.u[0] = *(const uint4*)(pb + 2 * bstep);  b2.u[1] = *(const uint4*)(pb + 2 * bstep + 16);
    b3.u[0] = *(const uint4*)(pb + 3 * bstep);  b3.u[1] = *(const uint4*)(pb + 3 * bstep + 16);
    __builtin_prefetch(pa0 + 32, 0, 3);         // WGP-scope global_prefetch_b8
    __builtin_prefetch(pa1 + 32, 0, 3);
    pa0 += 32; pa1 += 32; pb += 32;
    c00 = __builtin_amdgcn_wmma_f32_16x16x32_bf16(false, a0.f, false, b0.f, (short)0, c00, false, false);
    c01 = __builtin_amdgcn_wmma_f32_16x16x32_bf16(false, a0.f, false, b1.f, (short)0, c01, false, false);
    c02 = __builtin_amdgcn_wmma_f32_16x16x32_bf16(false, a0.f, false, b2.f, (short)0, c02, false, false);
    c03 = __builtin_amdgcn_wmma_f32_16x16x32_bf16(false, a0.f, false, b3.f, (short)0, c03, false, false);
    c10 = __builtin_amdgcn_wmma_f32_16x16x32_bf16(false, a1.f, false, b0.f, (short)0, c10, false, false);
    c11 = __builtin_amdgcn_wmma_f32_16x16x32_bf16(false, a1.f, false, b1.f, (short)0, c11, false, false);
    c12 = __builtin_amdgcn_wmma_f32_16x16x32_bf16(false, a1.f, false, b2.f, (short)0, c12, false, false);
    c13 = __builtin_amdgcn_wmma_f32_16x16x32_bf16(false, a1.f, false, b3.f, (short)0, c13, false, false);
  }
  int col = lane & 15;
  float w0 = bias[n0 + col], w1 = bias[n0 + col + 16],
        w2 = bias[n0 + col + 32], w3 = bias[n0 + col + 48];
  int rb0 = m0 + ((lane >> 4) << 3);      // D: VGPR r -> M=r (lanes 0-15) / r+8 (16-31)
  for (int r = 0; r < 8; ++r) {
    float* o = out + (size_t)(rb0 + r) * Ncols + n0 + col;
    o[0]  = c00[r] + w0;
    o[16] = c01[r] + w1;
    o[32] = c02[r] + w2;
    o[48] = c03[r] + w3;
  }
  int rb1 = rb0 + 16;
  for (int r = 0; r < 8; ++r) {
    float* o = out + (size_t)(rb1 + r) * Ncols + n0 + col;
    o[0]  = c10[r] + w0;
    o[16] = c11[r] + w1;
    o[32] = c12[r] + w2;
    o[48] = c13[r] + w3;
  }
}

// ---- edge score: logits[e,h] = att_h . leakyrelu(xl[src]+xr[dst]) + seg-max
__global__ void k_score(const float* __restrict__ xl, const float* __restrict__ xr,
                        const float* __restrict__ att, const int* __restrict__ ei,
                        int E0, int Etot, int Hh, int ld,
                        float* __restrict__ logits, unsigned* __restrict__ mmax) {
  int wid  = blockIdx.x * WPB + (int)(threadIdx.x >> 5);
  int lane = (int)(threadIdx.x & 31);
  if (wid >= Etot * Hh) return;
  int e = wid / Hh, h = wid - e * Hh;
  int src, dst;
  if (e < E0) { src = ei[e]; dst = ei[E0 + e]; } else { src = dst = e - E0; }
  int c = h * 128 + (lane << 2);
  float4 a = *(const float4*)(xl + (size_t)src * ld + c);
  float4 b = *(const float4*)(xr + (size_t)dst * ld + c);
  float4 w = *(const float4*)(att + h * 128 + (lane << 2));
  float s = 0.f, t;
  t = a.x + b.x; s += w.x * (t > 0.f ? t : 0.2f * t);
  t = a.y + b.y; s += w.y * (t > 0.f ? t : 0.2f * t);
  t = a.z + b.z; s += w.z * (t > 0.f ? t : 0.2f * t);
  t = a.w + b.w; s += w.w * (t > 0.f ? t : 0.2f * t);
  for (int off = 16; off; off >>= 1) s += __shfl_xor(s, off, 32);
  if (lane == 0) {
    logits[wid] = s;
    atomicMax(mmax + (size_t)dst * Hh + h, encf(s));
  }
}

__global__ void k_expsum(const int* __restrict__ ei, int E0, int Etot, int Hh,
                         float* __restrict__ logits, const unsigned* __restrict__ mmax,
                         float* __restrict__ denom) {
  int i = blockIdx.x * blockDim.x + threadIdx.x;
  if (i >= Etot * Hh) return;
  int e = i / Hh, h = i - e * Hh;
  int dst = (e < E0) ? ei[E0 + e] : (e - E0);
  float v = expf(logits[i] - decf(mmax[(size_t)dst * Hh + h]));
  logits[i] = v;                                      // reuse buffer as ev
  unsafeAtomicAdd(denom + (size_t)dst * Hh + h, v);   // global_atomic_add_f32
}

__global__ void k_scatter(const float* __restrict__ xl, const int* __restrict__ ei,
                          int E0, int Etot, int Hh, int ld,
                          const float* __restrict__ ev, const float* __restrict__ denom,
                          float* __restrict__ out) {
  int wid  = blockIdx.x * WPB + (int)(threadIdx.x >> 5);
  int lane = (int)(threadIdx.x & 31);
  if (wid >= Etot * Hh) return;
  int e = wid / Hh, h = wid - e * Hh;
  int src, dst;
  if (e < E0) { src = ei[e]; dst = ei[E0 + e]; } else { src = dst = e - E0; }
  float alpha = ev[wid] / (denom[(size_t)dst * Hh + h] + 1e-16f);
  int c = h * 128 + (lane << 2);
  float4 xs = *(const float4*)(xl + (size_t)src * ld + c);
  float* o = out + (size_t)dst * ld + c;
  unsafeAtomicAdd(o + 0, alpha * xs.x);
  unsafeAtomicAdd(o + 1, alpha * xs.y);
  unsafeAtomicAdd(o + 2, alpha * xs.z);
  unsafeAtomicAdd(o + 3, alpha * xs.w);
}

// ---------------------------------------------------------------------------
extern "C" void kernel_launch(void* const* d_in, const int* in_sizes, int n_in,
                              void* d_out, int out_size, void* d_ws, size_t ws_size,
                              hipStream_t stream) {
  const int IN = 77, H = 4, HC = 512, K1 = 96;
  const float* x   = (const float*)d_in[0];
  const int*   ei  = (const int*)d_in[1];
  const float *Wl1 = (const float*)d_in[2],  *bl1 = (const float*)d_in[3],
              *Wr1 = (const float*)d_in[4],  *br1 = (const float*)d_in[5],
              *at1 = (const float*)d_in[6],  *b1  = (const float*)d_in[7],
              *Wl2 = (const float*)d_in[8],  *bl2 = (const float*)d_in[9],
              *Wr2 = (const float*)d_in[10], *br2 = (const float*)d_in[11],
              *at2 = (const float*)d_in[12], *b2  = (const float*)d_in[13],
              *Wl3 = (const float*)d_in[14], *bl3 = (const float*)d_in[15],
              *Wr3 = (const float*)d_in[16], *br3 = (const float*)d_in[17],
              *at3 = (const float*)d_in[18], *b3  = (const float*)d_in[19];
  const int N    = in_sizes[0] / IN;        // 10000
  const int E0   = in_sizes[1] / 2;         // 160000
  const int Etot = E0 + N;                  // with self-loops
  const int Mpad = (N + 31) & ~31;          // 10016, multiple of 32
  const int mtiles = Mpad / 32;             // 32-row tiles

  char* wsp = (char*)d_ws; size_t off = 0;
  auto carve = [&](size_t bytes) { void* p = wsp + off; off = (off + bytes + 255) & ~(size_t)255; return p; };
  unsigned short* Abf = (unsigned short*)carve((size_t)Mpad * HC * 2);
  unsigned short* Blb = (unsigned short*)carve((size_t)HC * HC * 2);
  unsigned short* Brb = (unsigned short*)carve((size_t)HC * HC * 2);
  float*    xl     = (float*)carve((size_t)Mpad * HC * 4);
  float*    xr     = (float*)carve((size_t)Mpad * HC * 4);
  float*    hout   = (float*)carve((size_t)N * HC * 4);
  float*    logits = (float*)carve((size_t)Etot * H * 4);
  unsigned* mmax   = (unsigned*)carve((size_t)N * H * 4);
  float*    denom  = (float*)carve((size_t)N * H * 4);

  const int TB = 256;
  auto nb = [](long long n, int b) { return (unsigned)((n + b - 1) / b); };
  const unsigned NEG_MAX = 0x00800000u;  // encf(-FLT_MAX)

  // ===== Layer 1: 77 -> 4x128 =====
  k_convert_pad<<<nb((long long)Mpad * K1, TB), TB, 0, stream>>>(x, Abf, Mpad, N, IN, K1);
  k_convert_pad<<<nb((long long)HC * K1, TB), TB, 0, stream>>>(Wl1, Blb, HC, HC, IN, K1);
  k_convert_pad<<<nb((long long)HC * K1, TB), TB, 0, stream>>>(Wr1, Brb, HC, HC, IN, K1);
  k_fill_f32<<<nb((long long)N * HC, TB), TB, 0, stream>>>(hout, 0.f, N * HC);
  k_fill_f32<<<nb(N * H, TB), TB, 0, stream>>>(denom, 0.f, N * H);
  k_fill_u32<<<nb(N * H, TB), TB, 0, stream>>>(mmax, NEG_MAX, N * H);
  k_gemm_wmma<<<nb((long long)2 * mtiles * (HC / 64), WPB), TB, 0, stream>>>(
      Abf, Blb, Brb, bl1, br1, xl, xr, K1, HC, mtiles);
  k_score<<<nb((long long)Etot * H, WPB), TB, 0, stream>>>(xl, xr, at1, ei, E0, Etot, H, HC, logits, mmax);
  k_expsum<<<nb((long long)Etot * H, TB), TB, 0, stream>>>(ei, E0, Etot, H, logits, mmax, denom);
  k_scatter<<<nb((long long)Etot * H, WPB), TB, 0, stream>>>(xl, ei, E0, Etot, H, HC, logits, denom, hout);
  k_post_elu<<<nb((long long)Mpad * HC, TB), TB, 0, stream>>>(hout, b1, Abf, N * HC, Mpad * HC, HC);

  // ===== Layer 2: 512 -> 4x128 =====
  k_convert_pad<<<nb((long long)HC * HC, TB), TB, 0, stream>>>(Wl2, Blb, HC, HC, HC, HC);
  k_convert_pad<<<nb((long long)HC * HC, TB), TB, 0, stream>>>(Wr2, Brb, HC, HC, HC, HC);
  k_fill_f32<<<nb((long long)N * HC, TB), TB, 0, stream>>>(hout, 0.f, N * HC);
  k_fill_f32<<<nb(N * H, TB), TB, 0, stream>>>(denom, 0.f, N * H);
  k_fill_u32<<<nb(N * H, TB), TB, 0, stream>>>(mmax, NEG_MAX, N * H);
  k_gemm_wmma<<<nb((long long)2 * mtiles * (HC / 64), WPB), TB, 0, stream>>>(
      Abf, Blb, Brb, bl2, br2, xl, xr, HC, HC, mtiles);
  k_score<<<nb((long long)Etot * H, WPB), TB, 0, stream>>>(xl, xr, at2, ei, E0, Etot, H, HC, logits, mmax);
  k_expsum<<<nb((long long)Etot * H, TB), TB, 0, stream>>>(ei, E0, Etot, H, logits, mmax, denom);
  k_scatter<<<nb((long long)Etot * H, WPB), TB, 0, stream>>>(xl, ei, E0, Etot, H, HC, logits, denom, hout);
  k_post_elu<<<nb((long long)Mpad * HC, TB), TB, 0, stream>>>(hout, b2, Abf, N * HC, Mpad * HC, HC);

  // ===== Layer 3: 512 -> 1x128, writes f32 d_out =====
  float* out = (float*)d_out;
  const int C3 = 128;
  k_convert_pad<<<nb((long long)C3 * HC, TB), TB, 0, stream>>>(Wl3, Blb, C3, C3, HC, HC);
  k_convert_pad<<<nb((long long)C3 * HC, TB), TB, 0, stream>>>(Wr3, Brb, C3, C3, HC, HC);
  k_fill_f32<<<nb(N, TB), TB, 0, stream>>>(denom, 0.f, N);
  k_fill_u32<<<nb(N, TB), TB, 0, stream>>>(mmax, NEG_MAX, N);
  k_bias_init<<<nb((long long)N * C3, TB), TB, 0, stream>>>(out, b3, N * C3, C3);
  k_gemm_wmma<<<nb((long long)2 * mtiles * (C3 / 64), WPB), TB, 0, stream>>>(
      Abf, Blb, Brb, bl3, br3, xl, xr, HC, C3, mtiles);
  k_score<<<nb((long long)Etot, WPB), TB, 0, stream>>>(xl, xr, at3, ei, E0, Etot, 1, C3, logits, mmax);
  k_expsum<<<nb((long long)Etot, TB), TB, 0, stream>>>(ei, E0, Etot, 1, logits, mmax, denom);
  k_scatter<<<nb((long long)Etot, WPB), TB, 0, stream>>>(xl, ei, E0, Etot, 1, C3, logits, denom, out);
}
